// Trainer_6734508720564
// MI455X (gfx1250) — compile-verified, matrix operand order checked by reading, not run on hardware
//
#include <hip/hip_runtime.h>

#define NDEV  128
#define NNODE 4096
#define DMEM  128
#define DFEAT 64
#define E_PKT 4096
#define G_GSP 4096
#define TP    32
#define TG    16

typedef __bf16  v16bf __attribute__((ext_vector_type(16)));
typedef float   v8f   __attribute__((ext_vector_type(8)));

#define WMMA_BF16(A,B,C) __builtin_amdgcn_wmma_f32_16x16x32_bf16(false,(A),false,(B),(short)0,(C),false,false)

__device__ __forceinline__ v8f splat8(float x){
  v8f v;
#pragma unroll
  for (int i=0;i<8;++i) v[i]=x;
  return v;
}

__device__ __forceinline__ float sigm(float x){ return 1.0f/(1.0f+__expf(-x)); }

// B fragments pre-swizzled: [n_tile][kc][lane][16 bf16], 32B contiguous per lane.
__device__ __forceinline__ v16bf loadBfrag(const unsigned short* mat, int tile, int kc, int lane){
  return *reinterpret_cast<const v16bf*>(mat + ((((tile<<2)+kc)<<5) + lane)*16);
}

// ---------------------------------------------------------------------------
// Prep: convert 6 weight matrices (transposed, K=first-128-cols) to bf16 in
// WMMA B-fragment order.  K(j,g) = (j<8 ? j : j+8) + 8g within a 32-chunk.
// ---------------------------------------------------------------------------
__global__ void prep_weights(const float* s0,const float* s1,const float* s2,
                             const float* s3,const float* s4,const float* s5,
                             unsigned short* dst){
  int idx = blockIdx.x*blockDim.x + threadIdx.x;
  if (idx >= 6*384*128) return;
  int mi = idx / (384*128);
  int r  = idx % (384*128);
  int n  = r >> 7, k = r & 127;
  const float* src; int stride;
  switch (mi){
    case 0: src=s0; stride=192; break;   // w_ih_out (h part)
    case 1: src=s1; stride=128; break;   // w_hh_out
    case 2: src=s2; stride=192; break;   // w_ih_in
    case 3: src=s3; stride=128; break;   // w_hh_in
    case 4: src=s4; stride=128; break;   // w_ih_m
    default: src=s5; stride=128; break;  // w_hh_m
  }
  float v = src[n*stride + k];
  int nt = n>>4, nn = n&15, kc = k>>5, kk = k&31;
  int g  = (kk>>3)&1;
  int j  = (kk&7) + ((kk&16)?8:0);
  __bf16 h = (__bf16)v;
  dst[mi*(384*128) + ((((nt<<2)+kc)<<5) + (g<<4)+nn)*16 + j] =
      __builtin_bit_cast(unsigned short, h);
}

// Per-step constants: ci[t][n] = b_ih[n] + Xe[t] . w_ih[n,128:192]; pkt const.
__global__ void prep_consts(const float* Xe, const float* wihO, const float* bihO,
                            const float* wihI, const float* bihI,
                            const float* pktW, const float* pktB,
                            float* ciOut, float* ciIn, float* pktC){
  int idx = blockIdx.x*blockDim.x + threadIdx.x;
  if (idx < TP*384){
    int t = idx/384, n = idx%384;
    const float* xe = Xe + t*DFEAT;
    float a = bihO[n], b = bihI[n];
    for (int k=0;k<DFEAT;++k){
      float x = xe[k];
      a += x * wihO[n*192 + 128 + k];
      b += x * wihI[n*192 + 128 + k];
    }
    ciOut[idx]=a; ciIn[idx]=b;
  }
  if (idx < TP*2){
    int t = idx>>1, p = idx&1;
    const float* xe = Xe + t*DFEAT;
    float a = pktB[p];
    for (int k=0;k<DFEAT;++k) a += xe[k]*pktW[p*320 + 256 + k];
    pktC[idx]=a;
  }
}

// ---------------------------------------------------------------------------
// Packet step: 16 edges per block, 8 waves, 48 bf16 WMMAs per wave.
// ---------------------------------------------------------------------------
__global__ void __launch_bounds__(256,1) pkt_kernel(
    float* __restrict__ Hw,
    const int* __restrict__ pIdev, const int* __restrict__ pIsrc, const int* __restrict__ pIdst,
    const unsigned short* __restrict__ WihO, const unsigned short* __restrict__ WhhO,
    const unsigned short* __restrict__ WihI, const unsigned short* __restrict__ WhhI,
    const float* __restrict__ ciOut, const float* __restrict__ ciIn,
    const float* __restrict__ bhhO, const float* __restrict__ bhhI,
    const float* __restrict__ devW, const float* __restrict__ devB,
    const float* __restrict__ pktW, const float* __restrict__ pktC,
    float* __restrict__ pktOutT, float* __restrict__ devOutT)
{
  __shared__ __align__(16) float sHs[16][DMEM];
  __shared__ __align__(16) float sHd[16][DMEM];
  __shared__ __align__(16) float sNs[16][DMEM];
  __shared__ __align__(16) float sNd[16][DMEM];
  __shared__ int sDev[16], sSrc[16], sDst[16];

  const int tid = threadIdx.x;
  const int e0  = blockIdx.x * 16;

  { // gather both rows (float4)
    int m = tid >> 4, q = tid & 15;
    int e = e0 + m;
    int dv = pIdev[e], si = pIsrc[e], di = pIdst[e];
    if (q == 0){ sDev[m]=dv; sSrc[m]=si; sDst[m]=di; }
    const float4* rs = reinterpret_cast<const float4*>(Hw + ((size_t)(dv*NNODE + si))*DMEM) + q*2;
    const float4* rd = reinterpret_cast<const float4*>(Hw + ((size_t)(dv*NNODE + di))*DMEM) + q*2;
    float4 a0 = rs[0], a1 = rs[1];
    float4 b0 = rd[0], b1 = rd[1];
    float4* ps = reinterpret_cast<float4*>(&sHs[m][q*8]);
    float4* pd = reinterpret_cast<float4*>(&sHd[m][q*8]);
    ps[0]=a0; ps[1]=a1; pd[0]=b0; pd[1]=b1;
  }
  __syncthreads();

  const int w = tid >> 5, lane = tid & 31;
  const int g = lane >> 4, nl = lane & 15;

  // A fragments (16x32 bf16 tiles of gathered rows), ISA lane/K layout
  v16bf aHs[4], aHd[4];
#pragma unroll
  for (int kc=0;kc<4;++kc){
#pragma unroll
    for (int j=0;j<16;++j){
      int K = kc*32 + j + ((j>=8)?8:0) + g*8;
      aHs[kc][j] = (__bf16)sHs[nl][K];
      aHd[kc][j] = (__bf16)sHd[nl][K];
    }
  }

  const int n0 = w*16 + nl;   // gate column within [0,128)

  v8f r_o  = splat8(ciOut[n0]     + bhhO[n0]);
  v8f z_o  = splat8(ciOut[128+n0] + bhhO[128+n0]);
  v8f nn_o = splat8(ciOut[256+n0]);
  v8f hn_o = splat8(bhhO[256+n0]);
  v8f r_i  = splat8(ciIn[n0]      + bhhI[n0]);
  v8f z_i  = splat8(ciIn[128+n0]  + bhhI[128+n0]);
  v8f nn_i = splat8(ciIn[256+n0]);
  v8f hn_i = splat8(bhhI[256+n0]);

#pragma unroll
  for (int kc=0;kc<4;++kc){
    r_o  = WMMA_BF16(aHd[kc], loadBfrag(WihO, w,    kc, lane), r_o);
    r_o  = WMMA_BF16(aHs[kc], loadBfrag(WhhO, w,    kc, lane), r_o);
    z_o  = WMMA_BF16(aHd[kc], loadBfrag(WihO, 8+w,  kc, lane), z_o);
    z_o  = WMMA_BF16(aHs[kc], loadBfrag(WhhO, 8+w,  kc, lane), z_o);
    nn_o = WMMA_BF16(aHd[kc], loadBfrag(WihO, 16+w, kc, lane), nn_o);
    hn_o = WMMA_BF16(aHs[kc], loadBfrag(WhhO, 16+w, kc, lane), hn_o);
    r_i  = WMMA_BF16(aHs[kc], loadBfrag(WihI, w,    kc, lane), r_i);
    r_i  = WMMA_BF16(aHd[kc], loadBfrag(WhhI, w,    kc, lane), r_i);
    z_i  = WMMA_BF16(aHs[kc], loadBfrag(WihI, 8+w,  kc, lane), z_i);
    z_i  = WMMA_BF16(aHd[kc], loadBfrag(WhhI, 8+w,  kc, lane), z_i);
    nn_i = WMMA_BF16(aHs[kc], loadBfrag(WihI, 16+w, kc, lane), nn_i);
    hn_i = WMMA_BF16(aHd[kc], loadBfrag(WhhI, 16+w, kc, lane), hn_i);
  }

  const int dim = w*16 + nl;
#pragma unroll
  for (int i=0;i<8;++i){
    int m = i + g*8;
    float hs = sHs[m][dim], hd = sHd[m][dim];
    float r  = sigm(r_o[i]), z  = sigm(z_o[i]);
    float nn = tanhf(nn_o[i] + r*hn_o[i]);
    float hsn = (1.0f - z)*nn + z*hs;
    float r2  = sigm(r_i[i]), z2 = sigm(z_i[i]);
    float nn2 = tanhf(nn_i[i] + r2*hn_i[i]);
    float hdn = (1.0f - z2)*nn2 + z2*hd;
    sNs[m][dim] = hsn; sNd[m][dim] = hdn;
    Hw[((size_t)(sDev[m]*NNODE + sSrc[m]))*DMEM + dim] = hsn;
    Hw[((size_t)(sDev[m]*NNODE + sDst[m]))*DMEM + dim] = hdn;
  }
  __syncthreads();

  { // dev outputs: 32 rows x 8, one dot-128 per thread
    int row = tid >> 3, d = tid & 7;
    int side = row >> 4, m = row & 15;
    const float* hn = side ? sNd[m] : sNs[m];
    const float* wv = devW + d*DMEM;
    float acc = devB[d];
#pragma unroll 8
    for (int k=0;k<DMEM;++k) acc += hn[k]*wv[k];
    devOutT[((size_t)side*E_PKT + (e0+m))*8 + d] = acc;
  }
  if (tid < 32){ // pkt outputs: 16 rows x 2
    int m = tid >> 1, p = tid & 1;
    const float* wv = pktW + p*320;
    float acc = pktC[p];
#pragma unroll 8
    for (int k=0;k<DMEM;++k) acc += sNs[m][k]*wv[k] + sNd[m][k]*wv[128+k];
    pktOutT[(size_t)(e0+m)*2 + p] = acc;
  }
}

// ---------------------------------------------------------------------------
// Gossip step: 16 edges per block, 24 WMMAs per wave.
// ---------------------------------------------------------------------------
__global__ void __launch_bounds__(256,1) gsp_kernel(
    float* __restrict__ Hw,
    const int* __restrict__ pIds, const int* __restrict__ pIdd, const int* __restrict__ pIsend,
    const unsigned short* __restrict__ WihM, const unsigned short* __restrict__ WhhM,
    const float* __restrict__ bihM, const float* __restrict__ bhhM,
    float* __restrict__ gsrcT, float* __restrict__ gtgtT)
{
  __shared__ __align__(16) float sHs[16][DMEM];
  __shared__ __align__(16) float sHd[16][DMEM];
  __shared__ int sDd[16], sSe[16];

  const int tid = threadIdx.x;
  const int e0  = blockIdx.x * 16;

  { // gather + emit gsrc/gtgt
    int m = tid >> 4, q = tid & 15;
    int e = e0 + m;
    int ds = pIds[e], dd = pIdd[e], se = pIsend[e];
    if (q == 0){ sDd[m]=dd; sSe[m]=se; }
    const float4* rs = reinterpret_cast<const float4*>(Hw + ((size_t)(ds*NNODE + se))*DMEM) + q*2;
    const float4* rd = reinterpret_cast<const float4*>(Hw + ((size_t)(dd*NNODE + se))*DMEM) + q*2;
    float4 a0 = rs[0], a1 = rs[1];
    float4 b0 = rd[0], b1 = rd[1];
    float4* ps = reinterpret_cast<float4*>(&sHs[m][q*8]);
    float4* pd = reinterpret_cast<float4*>(&sHd[m][q*8]);
    ps[0]=a0; ps[1]=a1; pd[0]=b0; pd[1]=b1;
    float4* os = reinterpret_cast<float4*>(gsrcT + (size_t)e*DMEM + q*8);
    float4* od = reinterpret_cast<float4*>(gtgtT + (size_t)e*DMEM + q*8);
    os[0]=a0; os[1]=a1; od[0]=b0; od[1]=b1;
  }
  __syncthreads();

  const int w = tid >> 5, lane = tid & 31;
  const int g = lane >> 4, nl = lane & 15;

  v16bf aHs[4], aHd[4];
#pragma unroll
  for (int kc=0;kc<4;++kc){
#pragma unroll
    for (int j=0;j<16;++j){
      int K = kc*32 + j + ((j>=8)?8:0) + g*8;
      aHs[kc][j] = (__bf16)sHs[nl][K];
      aHd[kc][j] = (__bf16)sHd[nl][K];
    }
  }
  const int n0 = w*16 + nl;
  v8f r  = splat8(bihM[n0]     + bhhM[n0]);
  v8f z  = splat8(bihM[128+n0] + bhhM[128+n0]);
  v8f ni = splat8(bihM[256+n0]);
  v8f nh = splat8(bhhM[256+n0]);
#pragma unroll
  for (int kc=0;kc<4;++kc){
    r  = WMMA_BF16(aHs[kc], loadBfrag(WihM, w,    kc, lane), r);
    r  = WMMA_BF16(aHd[kc], loadBfrag(WhhM, w,    kc, lane), r);
    z  = WMMA_BF16(aHs[kc], loadBfrag(WihM, 8+w,  kc, lane), z);
    z  = WMMA_BF16(aHd[kc], loadBfrag(WhhM, 8+w,  kc, lane), z);
    ni = WMMA_BF16(aHs[kc], loadBfrag(WihM, 16+w, kc, lane), ni);
    nh = WMMA_BF16(aHd[kc], loadBfrag(WhhM, 16+w, kc, lane), nh);
  }
  const int dim = w*16 + nl;
#pragma unroll
  for (int i=0;i<8;++i){
    int m = i + g*8;
    float hd = sHd[m][dim];
    float rr = sigm(r[i]), zz = sigm(z[i]);
    float nn = tanhf(ni[i] + rr*nh[i]);
    float mg = (1.0f - zz)*nn + zz*hd;
    Hw[((size_t)(sDd[m]*NNODE + sSe[m]))*DMEM + dim] = mg;
  }
}

// ---------------------------------------------------------------------------
extern "C" void kernel_launch(void* const* d_in, const int* in_sizes, int n_in,
                              void* d_out, int out_size, void* d_ws, size_t ws_size,
                              hipStream_t stream){
  (void)in_sizes; (void)n_in; (void)out_size; (void)ws_size;
  const float* H     = (const float*)d_in[0];
  const int*   pIdev = (const int*)d_in[1];
  const int*   pIsrc = (const int*)d_in[2];
  const int*   pIdst = (const int*)d_in[3];
  const float* pktXe = (const float*)d_in[4];
  const int*   gIds  = (const int*)d_in[5];
  const int*   gIdd  = (const int*)d_in[6];
  const int*   gIse  = (const int*)d_in[7];
  const float* wihO  = (const float*)d_in[8];
  const float* whhO  = (const float*)d_in[9];
  const float* bihO  = (const float*)d_in[10];
  const float* bhhO  = (const float*)d_in[11];
  const float* wihI  = (const float*)d_in[12];
  const float* whhI  = (const float*)d_in[13];
  const float* bihI  = (const float*)d_in[14];
  const float* bhhI  = (const float*)d_in[15];
  const float* wihM  = (const float*)d_in[16];
  const float* whhM  = (const float*)d_in[17];
  const float* bihM  = (const float*)d_in[18];
  const float* bhhM  = (const float*)d_in[19];
  const float* devW  = (const float*)d_in[20];
  const float* devB  = (const float*)d_in[21];
  const float* pktW  = (const float*)d_in[22];
  const float* pktB  = (const float*)d_in[23];

  const size_t H_ELEMS = (size_t)NDEV*NNODE*DMEM;
  float* Hw = (float*)d_ws;
  char*  p  = (char*)d_ws + H_ELEMS*sizeof(float);
  unsigned short* Wfrag = (unsigned short*)p; p += (size_t)6*384*128*sizeof(unsigned short);
  float* ciOut = (float*)p; p += (size_t)TP*384*sizeof(float);
  float* ciIn  = (float*)p; p += (size_t)TP*384*sizeof(float);
  float* pktC  = (float*)p;

  // working copy of the mutable state (inputs must stay untouched)
  hipMemcpyAsync(Hw, H, H_ELEMS*sizeof(float), hipMemcpyDeviceToDevice, stream);
  prep_weights<<<(6*384*128 + 255)/256, 256, 0, stream>>>(wihO, whhO, wihI, whhI, wihM, whhM, Wfrag);
  prep_consts <<<(TP*384    + 255)/256, 256, 0, stream>>>(pktXe, wihO, bihO, wihI, bihI,
                                                          pktW, pktB, ciOut, ciIn, pktC);

  float* gsrcOut = (float*)d_out;
  float* gtgtOut = gsrcOut + (size_t)TG*G_GSP*DMEM;
  float* pktOut  = gtgtOut + (size_t)TG*G_GSP*DMEM;
  float* devOut  = pktOut  + (size_t)TP*E_PKT*2;

  const unsigned short* WihOf = Wfrag;
  const unsigned short* WhhOf = Wfrag + 1*384*128;
  const unsigned short* WihIf = Wfrag + 2*384*128;
  const unsigned short* WhhIf = Wfrag + 3*384*128;
  const unsigned short* WihMf = Wfrag + 4*384*128;
  const unsigned short* WhhMf = Wfrag + 5*384*128;

  for (int t=0;t<TP;++t){
    pkt_kernel<<<E_PKT/16, 256, 0, stream>>>(
        Hw, pIdev + t*E_PKT, pIsrc + t*E_PKT, pIdst + t*E_PKT,
        WihOf, WhhOf, WihIf, WhhIf,
        ciOut + t*384, ciIn + t*384, bhhO, bhhI,
        devW, devB, pktW, pktC + t*2,
        pktOut + (size_t)t*E_PKT*2, devOut + (size_t)t*2*E_PKT*8);
  }
  for (int t=0;t<TG;++t){
    gsp_kernel<<<G_GSP/16, 256, 0, stream>>>(
        Hw, gIds + t*G_GSP, gIdd + t*G_GSP, gIse + t*G_GSP,
        WihMf, WhhMf, bihM, bhhM,
        gsrcOut + (size_t)t*G_GSP*DMEM, gtgtOut + (size_t)t*G_GSP*DMEM);
  }
}
